// Phi3Mamba_63934883168593
// MI455X (gfx1250) — compile-verified, matrix operand order checked by reading, not run on hardware
//
#include <hip/hip_runtime.h>
#include <hip/hip_bf16.h>
#include <math.h>

#define D_MODEL 2048
#define D_STATE 16
#define D_CONV 4
#define D_INNER 4096
#define DT_RANK 128
#define B_SZ 2
#define SEQ_L 1024
#define M_ROWS (B_SZ * SEQ_L)          // 2048 token rows
#define N_SSM (DT_RANK + 2 * D_STATE)  // 160

#define USE_ASYNC_COPY 1               // gfx1250 GLOBAL_LOAD_ASYNC_TO_LDS path

typedef __attribute__((ext_vector_type(16))) __bf16 v16bf;
typedef __attribute__((ext_vector_type(8)))  __bf16 v8bf;
typedef __attribute__((ext_vector_type(4)))  __bf16 v4bf;
typedef __attribute__((ext_vector_type(8)))  float  v8f;

// Async 16B memory->LDS copy (per-lane), tracked by ASYNCcnt.
// LDS flat addresses have the byte offset in the low 32 bits (aperture layout),
// so truncating a generic pointer yields the VDST lds-offset operand.
__device__ __forceinline__ void async_copy_b128(void* lds_ptr, const void* gptr) {
#if USE_ASYNC_COPY
    unsigned int  lds = (unsigned int)(unsigned long long)lds_ptr;
    unsigned long long g = (unsigned long long)gptr;
    asm volatile("global_load_async_to_lds_b128 %0, %1, off"
                 :: "v"(lds), "v"(g) : "memory");
#else
    *(v8bf*)lds_ptr = *(const v8bf*)gptr;
#endif
}

__device__ __forceinline__ void wait_async() {
#if USE_ASYNC_COPY
    asm volatile("s_wait_asynccnt 0x0" ::: "memory");
#endif
}

// Assemble a v16bf WMMA fragment from LDS: elements {k0..k0+7, k0+16..k0+23}.
__device__ __forceinline__ v16bf ld_frag(const __bf16* p) {
    const v8bf lo = *(const v8bf*)p;
    const v8bf hi = *(const v8bf*)(p + 16);
    return __builtin_shufflevector(lo, hi,
        0, 1, 2, 3, 4, 5, 6, 7, 8, 9, 10, 11, 12, 13, 14, 15);
}

// ---------------------------------------------------------------------------
// f32 -> bf16 bulk converter.
// ---------------------------------------------------------------------------
__global__ __launch_bounds__(256)
void cvt_f32_to_bf16(const float* __restrict__ in, __bf16* __restrict__ out, int n4) {
    const int i = blockIdx.x * 256 + threadIdx.x;
    if (i < n4) {
        float4 f = ((const float4*)in)[i];
        v4bf o;
        o[0] = (__bf16)f.x; o[1] = (__bf16)f.y;
        o[2] = (__bf16)f.z; o[3] = (__bf16)f.w;
        ((v4bf*)out)[i] = o;
    }
}

// ---------------------------------------------------------------------------
// C[M,N] = A[M,K] * W[N,K]^T, A/W bf16, C f32 (+ optional bf16 mirror C2).
// MODE 0: plain. MODE 1: softplus(x + bias[n]) (dt_proj).
// Block = 256 threads = 8 wave32. Output tile 128(M) x 128(N).
// Double-buffered LDS tiles (A 128x32, B 128x32 per buffer, 32 KB total)
// filled with async memory->LDS copies; each wave owns a 16-row strip and
// issues 8 back-to-back WMMAs per K-step sharing one A fragment.
// N-edge handled by clamping (never-stored columns may hold garbage).
// ---------------------------------------------------------------------------
template <int MODE>
__global__ __launch_bounds__(256)
void gemm_bf16_wmma(const __bf16* __restrict__ A, int lda,
                    const __bf16* __restrict__ W, int ldb,
                    float* __restrict__ C, int ldc,
                    __bf16* __restrict__ C2,
                    int M, int N, int K,
                    const float* __restrict__ bias)
{
    __shared__ __align__(16) __bf16 sA[2][128 * 32];
    __shared__ __align__(16) __bf16 sB[2][128 * 32];

    const int tid  = threadIdx.x;
    const int lane = tid & 31;
    const int wave = tid >> 5;
    const int l16  = lane & 15;
    const int kgrp = lane >> 4;              // which K half-pack this lane holds
    const int mBase = blockIdx.y * 128;
    const int nBase = blockIdx.x * 128;

    // Fill addressing: 16B chunks; each thread moves 2 A-chunks + 2 B-chunks.
    const int row0 = tid >> 2;               // rows 0..63
    const int row1 = row0 + 64;              // rows 64..127
    const int cKo  = (tid & 3) * 8;
    const int aM0 = mBase + row0, aM1 = mBase + row1;
    const int bN0r = nBase + row0, bN1r = nBase + row1;
    const int bN0 = (bN0r < N) ? bN0r : (N - 1);   // clamp: branch-free edge
    const int bN1 = (bN1r < N) ? bN1r : (N - 1);

    v8f acc[8];
    #pragma unroll
    for (int j = 0; j < 8; ++j)
        #pragma unroll
        for (int e = 0; e < 8; ++e) acc[j][e] = 0.0f;

    // Prime buffer 0 with the first K tile.
    async_copy_b128(&sA[0][row0 * 32 + cKo], &A[(size_t)aM0 * lda + cKo]);
    async_copy_b128(&sA[0][row1 * 32 + cKo], &A[(size_t)aM1 * lda + cKo]);
    async_copy_b128(&sB[0][row0 * 32 + cKo], &W[(size_t)bN0 * ldb + cKo]);
    async_copy_b128(&sB[0][row1 * 32 + cKo], &W[(size_t)bN1 * ldb + cKo]);
    wait_async();
    __syncthreads();

    int cur = 0;
    for (int kb = 0; kb < K; kb += 32) {
        // Kick off async fills for the next K tile into the other buffer.
        const int nxt = cur ^ 1;
        if (kb + 32 < K) {
            const int kn = kb + 32;
            async_copy_b128(&sA[nxt][row0 * 32 + cKo], &A[(size_t)aM0 * lda + kn + cKo]);
            async_copy_b128(&sA[nxt][row1 * 32 + cKo], &A[(size_t)aM1 * lda + kn + cKo]);
            async_copy_b128(&sB[nxt][row0 * 32 + cKo], &W[(size_t)bN0 * ldb + kn + cKo]);
            async_copy_b128(&sB[nxt][row1 * 32 + cKo], &W[(size_t)bN1 * ldb + kn + cKo]);
        }

        // Preload ALL fragments, then issue the 8 WMMAs back-to-back.
        const int ko = kgrp * 8;
        const v16bf a = ld_frag(&sA[cur][(wave * 16 + l16) * 32 + ko]);
        v16bf bf[8];
        #pragma unroll
        for (int j = 0; j < 8; ++j)
            bf[j] = ld_frag(&sB[cur][(j * 16 + l16) * 32 + ko]);
        #pragma unroll
        for (int j = 0; j < 8; ++j)
            acc[j] = __builtin_amdgcn_wmma_f32_16x16x32_bf16(
                false, a, false, bf[j], (short)0, acc[j], false, false);

        // Next tile fully in LDS + everyone done reading the current one.
        wait_async();
        __syncthreads();
        cur = nxt;
    }

    // ---- epilogue: VGPR r -> M = r (+8 for lanes 16..31), N = lane&15 ----
    const int mOut = mBase + wave * 16 + kgrp * 8;
    #pragma unroll
    for (int j = 0; j < 8; ++j) {
        const int n = nBase + j * 16 + l16;
        if (n >= N) continue;
        #pragma unroll
        for (int r = 0; r < 8; ++r) {
            const int row = mOut + r;
            float v = acc[j][r];
            if (MODE == 1) {
                float t = v + bias[n];
                v = (t > 20.0f) ? t : log1pf(__expf(t));   // softplus
            }
            C[(size_t)row * ldc + n] = v;
            if (C2) C2[(size_t)row * ldc + n] = (__bf16)v;
        }
    }
}

// ---------------------------------------------------------------------------
// Depthwise causal conv1d (width 4) + bias + SiLU -> f32 (scan) + bf16 (GEMM).
// ---------------------------------------------------------------------------
__global__ __launch_bounds__(256)
void conv_silu_kernel(const float* __restrict__ proj,
                      const float* __restrict__ conv_w,
                      const float* __restrict__ conv_b,
                      float* __restrict__ xt,
                      __bf16* __restrict__ xt_bf)
{
    const int idx = blockIdx.x * 256 + threadIdx.x;    // (b,l,d) flattened
    const int d = idx & (D_INNER - 1);
    const int l = (idx >> 12) & (SEQ_L - 1);
    const int b = idx >> 22;
    float acc = conv_b[d];
    #pragma unroll
    for (int j = 0; j < D_CONV; ++j) {
        const int t = l - (D_CONV - 1) + j;
        if (t >= 0)
            acc += proj[((size_t)(b * SEQ_L + t)) * (2 * D_INNER) + d] * conv_w[d * D_CONV + j];
    }
    const float s = acc / (1.0f + __expf(-acc));       // SiLU
    xt[(size_t)idx]    = s;
    xt_bf[(size_t)idx] = (__bf16)s;
}

// ---------------------------------------------------------------------------
// Selective scan: one thread per channel (b,d); 16-wide state in registers;
// per-timestep B_t/C_t staged in LDS. Fuses D-skip + SiLU gating; emits bf16.
// ---------------------------------------------------------------------------
__global__ __launch_bounds__(256)
void scan_kernel(const float* __restrict__ dt,
                 const float* __restrict__ xt,
                 const float* __restrict__ ssm,
                 const float* __restrict__ proj,
                 const float* __restrict__ Amat,
                 const float* __restrict__ Dvec,
                 __bf16* __restrict__ ybuf)
{
    __shared__ float sBC[32];                          // [0..15]=B_t, [16..31]=C_t
    const int b = blockIdx.y;
    const int d = blockIdx.x * 256 + threadIdx.x;

    float a[D_STATE], st[D_STATE];
    #pragma unroll
    for (int s = 0; s < D_STATE; ++s) {
        a[s]  = Amat[d * D_STATE + s];
        st[s] = 0.0f;
    }
    const float Dv = Dvec[d];

    for (int l = 0; l < SEQ_L; ++l) {
        const size_t rowi = (size_t)(b * SEQ_L + l);
        __syncthreads();
        if (threadIdx.x < 32)
            sBC[threadIdx.x] = ssm[rowi * N_SSM + DT_RANK + threadIdx.x];
        __syncthreads();

        const float dtv = dt[rowi * D_INNER + d];      // already softplus'ed
        const float xv  = xt[rowi * D_INNER + d];
        float y = 0.0f;
        #pragma unroll
        for (int s = 0; s < D_STATE; ++s) {
            st[s] = st[s] * __expf(dtv * a[s]) + dtv * sBC[s] * xv;
            y += st[s] * sBC[D_STATE + s];
        }
        const float g = proj[rowi * (2 * D_INNER) + D_INNER + d];
        ybuf[rowi * D_INNER + d] = (__bf16)((y + xv * Dv) * (g / (1.0f + __expf(-g))));
    }
}

// ---------------------------------------------------------------------------
extern "C" void kernel_launch(void* const* d_in, const int* in_sizes, int n_in,
                              void* d_out, int out_size, void* d_ws, size_t ws_size,
                              hipStream_t stream) {
    const float* hidden     = (const float*)d_in[0];   // [B,L,2048]
    const float* in_proj_w  = (const float*)d_in[1];   // [8192,2048]
    const float* conv_w     = (const float*)d_in[2];   // [4096,4]
    const float* conv_b     = (const float*)d_in[3];   // [4096]
    const float* x_proj_w   = (const float*)d_in[4];   // [160,4096]
    const float* dt_proj_w  = (const float*)d_in[5];   // [4096,128]
    const float* dt_proj_b  = (const float*)d_in[6];   // [4096]
    const float* Amat       = (const float*)d_in[7];   // [4096,16]
    const float* Dvec       = (const float*)d_in[8];   // [4096]
    const float* out_proj_w = (const float*)d_in[9];   // [2048,4096]
    float* out = (float*)d_out;                        // [B,L,2048]

    // ---- workspace partition ----
    char* p = (char*)d_ws;
    auto alloc_bytes = [&](size_t nbytes) {
        void* r = (void*)p;
        p += (nbytes + 255) & ~(size_t)255;
        return r;
    };
    float*  proj   = (float*) alloc_bytes((size_t)M_ROWS * 2 * D_INNER * 4); // 64 MB
    float*  xt     = (float*) alloc_bytes((size_t)M_ROWS * D_INNER * 4);     // 32 MB
    float*  ssm    = (float*) alloc_bytes((size_t)M_ROWS * N_SSM * 4);       // 1.25 MB
    float*  dtb    = (float*) alloc_bytes((size_t)M_ROWS * D_INNER * 4);     // 32 MB
    __bf16* hid_b  = (__bf16*)alloc_bytes((size_t)M_ROWS * D_MODEL * 2);     // 8 MB
    __bf16* inW_b  = (__bf16*)alloc_bytes((size_t)2 * D_INNER * D_MODEL * 2);// 32 MB
    __bf16* xW_b   = (__bf16*)alloc_bytes((size_t)N_SSM * D_INNER * 2);      // 1.25 MB
    __bf16* dtW_b  = (__bf16*)alloc_bytes((size_t)D_INNER * DT_RANK * 2);    // 1 MB
    __bf16* outW_b = (__bf16*)alloc_bytes((size_t)D_MODEL * D_INNER * 2);    // 16 MB
    __bf16* xt_b   = (__bf16*)alloc_bytes((size_t)M_ROWS * D_INNER * 2);     // 16 MB
    __bf16* ssm_b  = (__bf16*)alloc_bytes((size_t)M_ROWS * N_SSM * 2);       // 0.6 MB
    __bf16* y_b    = (__bf16*)alloc_bytes((size_t)M_ROWS * D_INNER * 2);     // 16 MB

    dim3 blk(256);
    auto cvt = [&](const float* src, __bf16* dst, size_t n) {
        const int n4 = (int)(n / 4);
        cvt_f32_to_bf16<<<(n4 + 255) / 256, blk, 0, stream>>>(src, dst, n4);
    };

    // 0) one-time bf16 conversions (weights + input activations)
    cvt(hidden,     hid_b,  (size_t)M_ROWS * D_MODEL);
    cvt(in_proj_w,  inW_b,  (size_t)2 * D_INNER * D_MODEL);
    cvt(x_proj_w,   xW_b,   (size_t)N_SSM * D_INNER);
    cvt(dt_proj_w,  dtW_b,  (size_t)D_INNER * DT_RANK);
    cvt(out_proj_w, outW_b, (size_t)D_MODEL * D_INNER);

    // 1) in_proj: [2048,2048] x [8192,2048]^T -> proj f32 [2048,8192]
    gemm_bf16_wmma<0><<<dim3((2 * D_INNER) / 128, M_ROWS / 128), blk, 0, stream>>>(
        hid_b, D_MODEL, inW_b, D_MODEL, proj, 2 * D_INNER, (__bf16*)nullptr,
        M_ROWS, 2 * D_INNER, D_MODEL, nullptr);

    // 2) depthwise causal conv + SiLU -> xt f32 + bf16
    conv_silu_kernel<<<(M_ROWS * D_INNER) / 256, blk, 0, stream>>>(
        proj, conv_w, conv_b, xt, xt_b);

    // 3) x_proj: [2048,4096] x [160,4096]^T -> ssm f32 + bf16 [2048,160]
    gemm_bf16_wmma<0><<<dim3((N_SSM + 127) / 128, M_ROWS / 128), blk, 0, stream>>>(
        xt_b, D_INNER, xW_b, D_INNER, ssm, N_SSM, ssm_b,
        M_ROWS, N_SSM, D_INNER, nullptr);

    // 4) dt_proj + softplus(x + bias): ssm[:, :128] x [4096,128]^T -> dtb f32
    gemm_bf16_wmma<1><<<dim3(D_INNER / 128, M_ROWS / 128), blk, 0, stream>>>(
        ssm_b, N_SSM, dtW_b, DT_RANK, dtb, D_INNER, (__bf16*)nullptr,
        M_ROWS, D_INNER, DT_RANK, dt_proj_b);

    // 5) selective scan (+ D skip + SiLU gate) -> y bf16 [2048,4096]
    scan_kernel<<<dim3(D_INNER / 256, B_SZ), blk, 0, stream>>>(
        dtb, xt, ssm, proj, Amat, Dvec, y_b);

    // 6) out_proj: [2048,4096] x [2048,4096]^T -> out f32 [2048,2048]
    gemm_bf16_wmma<0><<<dim3(D_MODEL / 128, M_ROWS / 128), blk, 0, stream>>>(
        y_b, D_INNER, outW_b, D_INNER, out, D_MODEL, (__bf16*)nullptr,
        M_ROWS, D_MODEL, D_INNER, nullptr);
}
